// BatchedNMS_1202590843778
// MI455X (gfx1250) — matrix operand order repaired and checked by reading, not compiled, not run on hardware
//
#include <hip/hip_runtime.h>

// ---------------- problem constants (fixed by reference setup) ----------------
#define BATCH   8
#define NBOX    20000
#define NCLS    80
#define TOPK    500
#define KEEPK   100
#define NBINS   128
#define CANDMAX 1024
#define SCORE_THR 0.05f
#define IOU_THR   0.5f
#define CHUNK_N 500
#define CHUNKS  (NBOX / CHUNK_N)   // 40

typedef __attribute__((ext_vector_type(16))) _Float16 v16h;
typedef __attribute__((ext_vector_type(8)))  float    v8f;

__device__ __forceinline__ int imin(int a, int b) { return a < b ? a : b; }

__device__ __forceinline__ float iou_pair(float ax1, float ay1, float ax2, float ay2, float aa,
                                          float bx1, float by1, float bx2, float by2, float ba) {
    float iw = fminf(ax2, bx2) - fmaxf(ax1, bx1);
    float ih = fminf(ay2, by2) - fmaxf(ay1, by1);
    iw = fmaxf(iw, 0.0f);
    ih = fmaxf(ih, 0.0f);
    float inter = iw * ih;
    float uni   = aa + ba - inter;
    return inter / fmaxf(uni, 1e-9f);
}

// ---------------- kernel 0: zero scratch ----------------
__global__ void bnms_zero_kernel(int* p, int n) {
    int i = blockIdx.x * blockDim.x + threadIdx.x;
    for (; i < n; i += gridDim.x * blockDim.x) p[i] = 0;
}

// ---------------- kernel 1: per-(b,class) score histograms ----------------
// Coalesced sweep over scores; LDS hist[80][128]; flush via global atomics.
__global__ __launch_bounds__(256) void bnms_hist_kernel(const float* __restrict__ scores,
                                                        int* __restrict__ hist) {
    __shared__ int lh[NCLS * NBINS];          // 40 KB
    const int wg    = blockIdx.x;
    const int b     = wg / CHUNKS;
    const int chunk = wg % CHUNKS;
    for (int i = threadIdx.x; i < NCLS * NBINS; i += blockDim.x) lh[i] = 0;
    __syncthreads();

    const float* base = scores + ((size_t)b * NBOX + (size_t)chunk * CHUNK_N) * NCLS;
    const int total = CHUNK_N * NCLS;         // 40000 floats, contiguous
    for (int f = threadIdx.x; f < total; f += blockDim.x) {
        if (f + 8192 < total) __builtin_prefetch(base + f + 8192, 0, 0);
        float s = base[f];
        if (s > SCORE_THR) {
            int c   = f % NCLS;
            int bin = imin(NBINS - 1, (int)(s * (float)NBINS));
            atomicAdd(&lh[c * NBINS + bin], 1);
        }
    }
    __syncthreads();

    int* gh = hist + (size_t)b * NCLS * NBINS;
    for (int i = threadIdx.x; i < NCLS * NBINS; i += blockDim.x)
        if (lh[i]) atomicAdd(&gh[i], lh[i]);
}

// ---------------- kernel 2: per-slot top-K + exact NMS with WMMA propagation ----------------
__global__ __launch_bounds__(512) void bnms_nms_kernel(const float* __restrict__ scores,
                                                       const float* __restrict__ bboxes,
                                                       const int* __restrict__ hist,
                                                       float* __restrict__ kept_s,
                                                       int*   __restrict__ kept_k,
                                                       int*   __restrict__ keptCount,
                                                       int*   __restrict__ keptTrue) {
    const int slot = blockIdx.x;
    const int b    = slot / NCLS;
    const int c    = slot % NCLS;
    const int tid  = threadIdx.x;

    __shared__ float cs[CANDMAX];
    __shared__ int   ci[CANDMAX];
    __shared__ float bx1[512], by1[512], bx2[512], by2[512], bar[512];
    __shared__ int   keep[512];
    __shared__ int   s_cnt, s_bin;

    // -- threshold bin from histogram (serial over 128 bins; trivial) --
    if (tid == 0) {
        const int* h = hist + (size_t)slot * NBINS;
        int cum = 0, bsel = 0;
        for (int bn = NBINS - 1; bn >= 0; --bn) {
            cum += h[bn];
            if (cum >= TOPK) { bsel = bn; break; }
        }
        s_bin = bsel;
        s_cnt = 0;
    }
    __syncthreads();
    const int binThr = s_bin;

    // -- collect candidates (column reads; L2-resident after kernel 1) --
    const float* colBase = scores + (size_t)b * NBOX * NCLS + c;
    for (int n = tid; n < NBOX; n += 512) {
        if (n + 2048 < NBOX) __builtin_prefetch(colBase + (size_t)(n + 2048) * NCLS, 0, 0);
        float s = colBase[(size_t)n * NCLS];
        if (s > SCORE_THR) {
            int bin = imin(NBINS - 1, (int)(s * (float)NBINS));
            if (bin >= binThr) {
                int p = atomicAdd(&s_cnt, 1);
                if (p < CANDMAX) { cs[p] = s; ci[p] = n; }
            }
        }
    }
    __syncthreads();
    const int m = imin(s_cnt, CANDMAX);
    for (int i = tid; i < CANDMAX; i += 512)
        if (i >= m) { cs[i] = -__builtin_inff(); ci[i] = 0; }
    __syncthreads();

    // -- bitonic sort 1024 entries, descending by score --
    for (int k = 2; k <= CANDMAX; k <<= 1) {
        for (int j = k >> 1; j > 0; j >>= 1) {
            for (int e = tid; e < CANDMAX; e += 512) {
                int x = e ^ j;
                if (x > e) {
                    bool segUp = ((e & k) == 0);
                    float v0 = cs[e], v1 = cs[x];
                    bool sw = segUp ? (v0 < v1) : (v0 > v1);   // descending overall
                    if (sw) {
                        cs[e] = v1; cs[x] = v0;
                        int t0 = ci[e]; ci[e] = ci[x]; ci[x] = t0;
                    }
                }
            }
            __syncthreads();
        }
    }

    // -- load boxes for the top min(m, TOPK); zero-fill invalid (keeps WMMA NaN-free) --
    const int limit = imin(m, TOPK);
    {
        int j = tid;              // 512 threads cover 512 slots
        bool valid = (j < limit);
        float x1 = 0.f, y1 = 0.f, x2 = 0.f, y2 = 0.f, ar = 0.f;
        if (valid) {
            const float* bb = bboxes + ((size_t)b * NBOX + ci[j]) * 4;
            x1 = bb[0]; y1 = bb[1]; x2 = bb[2]; y2 = bb[3];
            ar = (x2 - x1) * (y2 - y1);
        }
        bx1[j] = x1; by1[j] = y1; bx2[j] = x2; by2[j] = y2; bar[j] = ar;
        keep[j] = valid ? 1 : 0;
    }
    __syncthreads();

    // -- exact NMS in 16-wide blocks; cross-block propagation via v_wmma --
    const int wave = tid >> 5;
    const int lane = tid & 31;
    const bool loLane = (lane < 16);
    for (int t = 0; t < 32; ++t) {
        // intra-block sequential resolve (wave 0, lanes 0..15 own columns)
        if (tid < 32) {
            const int jl = t * 16 + (lane & 15);
            const bool act = lane < 16;
            float jx1 = bx1[jl], jy1 = by1[jl], jx2 = bx2[jl], jy2 = by2[jl], ja = bar[jl];
            int kp = act ? keep[jl] : 0;
#pragma unroll
            for (int i = 0; i < 16; ++i) {
                int ki = __shfl(kp, i);   // suppressor's current resolved keep
                if (act && (lane & 15) > i && ki) {
                    int gi = t * 16 + i;
                    float v = iou_pair(bx1[gi], by1[gi], bx2[gi], by2[gi], bar[gi],
                                       jx1, jy1, jx2, jy2, ja);
                    if (v > IOU_THR) kp = 0;
                }
            }
            if (act) keep[jl] = kp;
        }
        __syncthreads();

        // A fragment: 16x32 f16, rows all equal to the block's keep mask (K<16), zeros K>=16.
        // 16-bit A layout (contiguous, matches B): lanes<16 -> K=0..15; lanes>=16 -> K=16..31.
        v16h a;
        {
            const int kb = t * 16;
#pragma unroll
            for (int v = 0; v < 8; ++v) {
                a[2 * v + 0] = loLane ? (_Float16)(float)keep[kb + 2 * v + 0] : (_Float16)0.0f;
                a[2 * v + 1] = loLane ? (_Float16)(float)keep[kb + 2 * v + 1] : (_Float16)0.0f;
            }
        }

        // Propagate to all later 16-column tiles; one tile per wave per pass.
        for (int u = t + 1 + wave; u < 32; u += 16) {
            const int j0 = u * 16;
            const int jj = j0 + (lane & 15);
            const float jx1 = bx1[jj], jy1 = by1[jj], jx2 = bx2[jj], jy2 = by2[jj], ja = bar[jj];
            // B fragment: 32x16 f16. Lanes<16 hold K=0..15 (2 per VGPR), lanes>=16 hold K>=16 -> 0.
            // Branchless per-element select; fully unrolled so the fragment stays in VGPRs
            // (a runtime element-insertion loop forces a scratch round-trip).
            v16h bf;
#pragma unroll
            for (int v = 0; v < 8; ++v) {
#pragma unroll
                for (int h = 0; h < 2; ++h) {
                    const int gi = t * 16 + 2 * v + h;
                    float vio = iou_pair(bx1[gi], by1[gi], bx2[gi], by2[gi], bar[gi],
                                         jx1, jy1, jx2, jy2, ja);
                    bool on = loLane & (vio > IOU_THR);
                    bf[2 * v + h] = on ? (_Float16)1.0f : (_Float16)0.0f;
                }
            }
            v8f acc = {};
            // D[m][n] = sum_k keep[k] * S[k][n]  (> 0 => column suppressed)
            acc = __builtin_amdgcn_wmma_f32_16x16x32_f16(
                /*neg_a=*/false, a, /*neg_b=*/false, bf,
                /*c_mod=*/(short)0, acc, /*reuse_a=*/false, /*reuse_b=*/false);
            // lanes 0..15 own columns (row M=0); lanes 16..31 carry identical column sums.
            if (loLane && acc[0] > 0.5f) keep[jj] = 0;
        }
        __syncthreads();
    }

    // -- ordered compaction of survivors (descending score order preserved) --
    if (tid < limit && keep[tid]) {
        int r = 0;
        for (int i = 0; i < tid; ++i) r += keep[i];
        if (r < KEEPK) {
            kept_s[(size_t)slot * KEEPK + r] = cs[tid];
            kept_k[(size_t)slot * KEEPK + r] = (c << 16) | ci[tid];
        }
    }
    if (tid == 0) {
        int tot = 0;
        for (int i = 0; i < limit; ++i) tot += keep[i];
        keptTrue[slot]  = tot;
        keptCount[slot] = imin(tot, KEEPK);
    }
}

// ---------------- kernel 3: per-batch merge -> top keepTopK, write outputs ----------------
__global__ __launch_bounds__(1024) void bnms_final_kernel(const float* __restrict__ bboxes,
                                                          const float* __restrict__ kept_s,
                                                          const int*   __restrict__ kept_k,
                                                          const int*   __restrict__ keptCount,
                                                          const int*   __restrict__ keptTrue,
                                                          float* __restrict__ out) {
    __shared__ float ls[8192];                 // 32 KB of candidate scores
    __shared__ float rmax[1024];
    __shared__ int   rarg[1024];
    const int b   = blockIdx.x;
    const int tid = threadIdx.x;

    for (int i = tid; i < 8192; i += 1024) ls[i] = -__builtin_inff();
    __syncthreads();
    for (int i = tid; i < NCLS * KEEPK; i += 1024) {
        int s = i / KEEPK, e = i % KEEPK;
        if (e < keptCount[b * NCLS + s]) ls[i] = kept_s[(size_t)(b * NCLS + s) * KEEPK + e];
    }
    __syncthreads();

    float* out_nd  = out;                               // (B,1)
    float* out_box = out + BATCH;                       // (B,KEEPK,4)
    float* out_sc  = out_box + BATCH * KEEPK * 4;       // (B,KEEPK)
    float* out_cl  = out_sc  + BATCH * KEEPK;           // (B,KEEPK)

    if (tid == 0) {
        int tot = 0;
        for (int s = 0; s < NCLS; ++s) tot += keptTrue[b * NCLS + s];
        out_nd[b] = (float)imin(tot, KEEPK);
    }

    for (int k = 0; k < KEEPK; ++k) {
        float best = -__builtin_inff();
        int   bi   = -1;
        for (int i = tid; i < 8192; i += 1024) {
            float v = ls[i];
            if (v > best) { best = v; bi = i; }
        }
        rmax[tid] = best; rarg[tid] = bi;
        __syncthreads();
        for (int off = 512; off > 0; off >>= 1) {
            if (tid < off) {
                if (rmax[tid + off] > rmax[tid]) {
                    rmax[tid] = rmax[tid + off];
                    rarg[tid] = rarg[tid + off];
                }
            }
            __syncthreads();
        }
        if (tid == 0) {
            float sv = rmax[0];
            int   wi = rarg[0];
            float sc = 0.0f, cl = -1.0f, x1 = 0.f, y1 = 0.f, x2 = 0.f, y2 = 0.f;
            if (wi >= 0 && sv > -__builtin_inff()) {
                int s   = wi / KEEPK, e = wi % KEEPK;
                int key = kept_k[(size_t)(b * NCLS + s) * KEEPK + e];
                int n   = key & 0xFFFF;
                cl = (float)(key >> 16);
                sc = sv;
                const float* bb = bboxes + ((size_t)b * NBOX + n) * 4;
                x1 = bb[0]; y1 = bb[1]; x2 = bb[2]; y2 = bb[3];
                ls[wi] = -__builtin_inff();
            }
            size_t o = (size_t)(b * KEEPK + k);
            out_sc[o] = sc;
            out_cl[o] = cl;
            out_box[o * 4 + 0] = x1;
            out_box[o * 4 + 1] = y1;
            out_box[o * 4 + 2] = x2;
            out_box[o * 4 + 3] = y2;
        }
        __syncthreads();
    }
}

// ---------------- launcher ----------------
extern "C" void kernel_launch(void* const* d_in, const int* in_sizes, int n_in,
                              void* d_out, int out_size, void* d_ws, size_t ws_size,
                              hipStream_t stream) {
    const float* scores = (const float*)d_in[0];
    const float* bboxes = (const float*)d_in[1];
    (void)in_sizes; (void)n_in; (void)out_size; (void)ws_size;

    // workspace layout (all 4-byte aligned): ~0.85 MB total
    int*   hist      = (int*)d_ws;                         // B*C*NBINS
    int*   keptCount = hist + BATCH * NCLS * NBINS;        // B*C
    int*   keptTrue  = keptCount + BATCH * NCLS;           // B*C
    float* kept_s    = (float*)(keptTrue + BATCH * NCLS);  // B*C*KEEPK
    int*   kept_k    = (int*)(kept_s + BATCH * NCLS * KEEPK);

    const int zn = BATCH * NCLS * NBINS + 2 * BATCH * NCLS;
    bnms_zero_kernel<<<(zn + 255) / 256, 256, 0, stream>>>(hist, zn);
    bnms_hist_kernel<<<BATCH * CHUNKS, 256, 0, stream>>>(scores, hist);
    bnms_nms_kernel<<<BATCH * NCLS, 512, 0, stream>>>(scores, bboxes, hist,
                                                      kept_s, kept_k, keptCount, keptTrue);
    bnms_final_kernel<<<BATCH, 1024, 0, stream>>>(bboxes, kept_s, kept_k,
                                                  keptCount, keptTrue, (float*)d_out);
}
